// SSMAttention_70703751627160
// MI455X (gfx1250) — compile-verified, matrix operand order checked by reading, not run on hardware
//
#include <hip/hip_runtime.h>
#include <hip/hip_bf16.h>
#include <math.h>

// ---------------------------------------------------------------------------
// Mamba (2 layers) + MLP for gfx1250 (MI455X).
// GEMMs: V_WMMA_F32_16X16X32_F16 (wave32), 2x2 fragments per wave,
//        double-buffered LDS tiles, batched b128 staging loads.
// Scan : 16 SSM states -> 16 lanes of a half-wave, shfl_xor(width=16) reduce.
// ---------------------------------------------------------------------------

typedef __attribute__((ext_vector_type(16))) _Float16 v16h;
typedef __attribute__((ext_vector_type(8)))  _Float16 v8h;
typedef __attribute__((ext_vector_type(4)))  _Float16 v4h;
typedef __attribute__((ext_vector_type(8)))  float    v8f;

#define D_MODEL 512
#define D_STATE 16
#define D_CONV  4
#define D_INNER 1024
#define DT_RANK 32
#define FF_DIM  1024
#define N_LAYERS 2
#define BB 2
#define LSEQ 2048
#define MM (BB * LSEQ)   // 4096 rows in every GEMM

#define EP_NONE          0
#define EP_BIAS          1
#define EP_BIAS_RELU     2
#define EP_BIAS_SOFTPLUS 3

// LDS row pitch in halfs: keeps 16B alignment, skews banks
#define APITCH 40

// ---------------------------------------------------------------------------
// WMMA GEMM: C[M,N] = A[M,K] * W[N,K]^T (+ bias / activation)
// Block = 256 threads (8 wave32s). Block tile: 128(M) x 64(N). K step 32.
// Wave (waveM = w&3, waveN = w>>2) -> 32x32 C tile = 2x2 WMMA fragments.
// Ping-pong LDS: loads for tile kt+1 are issued before the WMMAs of tile kt.
// Requires: M % 128 == 0, N % 64 == 0, K % 32 == 0.
// ---------------------------------------------------------------------------
__global__ __launch_bounds__(256) void gemm_wmma_f16(
    const float* __restrict__ A, const float* __restrict__ W,
    float* __restrict__ C, const float* __restrict__ bias,
    int M, int N, int K, int lda, int ldb, int ldc, int ep)
{
  __shared__ __align__(16) _Float16 sA[2][128][APITCH];
  __shared__ __align__(16) _Float16 sW[2][64][APITCH];

  const int t     = threadIdx.x;
  const int lane  = t & 31;           // wave32
  const int wave  = t >> 5;
  const int half  = lane >> 4;        // K-half select per ISA 16-bit layout
  const int l16   = lane & 15;
  const int rowBase = blockIdx.x * 128;
  const int colBase = blockIdx.y * 64;
  const int waveM = (wave & 3) * 32;  // within-block M offset (32 rows)
  const int waveN = (wave >> 2) * 32; // within-block N offset (32 cols)

  // Per-thread staging coordinates: float4 chunks over the K-tile
  // A: 128x32 = 1024 chunks -> 4/thread ; W: 64x32 = 512 chunks -> 2/thread
  int rAs[4], cAs[4], rWs[2], cWs[2];
#pragma unroll
  for (int i = 0; i < 4; ++i) {
    int idx = t + i * 256;
    rAs[i] = idx >> 3; cAs[i] = (idx & 7) * 4;
  }
#pragma unroll
  for (int i = 0; i < 2; ++i) {
    int idx = t + i * 256;
    rWs[i] = idx >> 3; cWs[i] = (idx & 7) * 4;
  }

  v8f acc[2][2] = {};
  const int kTiles = K >> 5;
  float4 fa[4], fw[2];

  // Prologue: load + stage K-tile 0 into buffer 0
#pragma unroll
  for (int i = 0; i < 4; ++i)
    fa[i] = *(const float4*)(A + (size_t)(rowBase + rAs[i]) * lda + cAs[i]);
#pragma unroll
  for (int i = 0; i < 2; ++i)
    fw[i] = *(const float4*)(W + (size_t)(colBase + rWs[i]) * ldb + cWs[i]);
#pragma unroll
  for (int i = 0; i < 4; ++i) {
    v4h h; h[0] = (_Float16)fa[i].x; h[1] = (_Float16)fa[i].y;
           h[2] = (_Float16)fa[i].z; h[3] = (_Float16)fa[i].w;
    *(v4h*)&sA[0][rAs[i]][cAs[i]] = h;
  }
#pragma unroll
  for (int i = 0; i < 2; ++i) {
    v4h h; h[0] = (_Float16)fw[i].x; h[1] = (_Float16)fw[i].y;
           h[2] = (_Float16)fw[i].z; h[3] = (_Float16)fw[i].w;
    *(v4h*)&sW[0][rWs[i]][cWs[i]] = h;
  }

  int cur = 0;
  for (int kt = 0; kt < kTiles; ++kt) {
    __syncthreads();                       // buf[cur] visible; buf[cur^1] free
    const bool hasNext = (kt + 1) < kTiles;
    const int kn = (kt + 1) << 5;

    // Issue next tile's global loads (in flight during the WMMAs below)
    if (hasNext) {
#pragma unroll
      for (int i = 0; i < 4; ++i)
        fa[i] = *(const float4*)(A + (size_t)(rowBase + rAs[i]) * lda + kn + cAs[i]);
#pragma unroll
      for (int i = 0; i < 2; ++i)
        fw[i] = *(const float4*)(W + (size_t)(colBase + rWs[i]) * ldb + kn + cWs[i]);
    }
    if (kt + 2 < kTiles) {                 // warm L2 for the tile after next
      __builtin_prefetch(&A[(size_t)(rowBase + (t >> 1)) * lda + kn + 32], 0, 1);
      __builtin_prefetch(&W[(size_t)(colBase + (t >> 2)) * ldb + kn + 32], 0, 1);
    }

    // Fragment gather per CDNA5 16-bit A/B layouts:
    // lane half 0 -> K = {0..7, 16..23}; half 1 -> K = {8..15, 24..31}
    v16h av[2], wv[2];
#pragma unroll
    for (int mi = 0; mi < 2; ++mi) {
      const _Float16* pa = &sA[cur][waveM + mi * 16 + l16][half * 8];
      v8h lo = *(const v8h*)pa;
      v8h hi = *(const v8h*)(pa + 16);
#pragma unroll
      for (int i = 0; i < 8; ++i) { av[mi][i] = lo[i]; av[mi][8 + i] = hi[i]; }
    }
#pragma unroll
    for (int ni = 0; ni < 2; ++ni) {
      const _Float16* pw = &sW[cur][waveN + ni * 16 + l16][half * 8];
      v8h lo = *(const v8h*)pw;
      v8h hi = *(const v8h*)(pw + 16);
#pragma unroll
      for (int i = 0; i < 8; ++i) { wv[ni][i] = lo[i]; wv[ni][8 + i] = hi[i]; }
    }
#pragma unroll
    for (int mi = 0; mi < 2; ++mi)
#pragma unroll
      for (int ni = 0; ni < 2; ++ni)
        acc[mi][ni] = __builtin_amdgcn_wmma_f32_16x16x32_f16(
            /*neg_a=*/false, av[mi], /*neg_b=*/false, wv[ni],
            /*c_mod=*/(short)0, acc[mi][ni],
            /*reuse_a=*/false, /*reuse_b=*/false);

    // Convert + stage next tile into the other buffer
    if (hasNext) {
      int nxt = cur ^ 1;
#pragma unroll
      for (int i = 0; i < 4; ++i) {
        v4h h; h[0] = (_Float16)fa[i].x; h[1] = (_Float16)fa[i].y;
               h[2] = (_Float16)fa[i].z; h[3] = (_Float16)fa[i].w;
        *(v4h*)&sA[nxt][rAs[i]][cAs[i]] = h;
      }
#pragma unroll
      for (int i = 0; i < 2; ++i) {
        v4h h; h[0] = (_Float16)fw[i].x; h[1] = (_Float16)fw[i].y;
               h[2] = (_Float16)fw[i].z; h[3] = (_Float16)fw[i].w;
        *(v4h*)&sW[nxt][rWs[i]][cWs[i]] = h;
      }
    }
    cur ^= 1;
  }

  // Store per 16x16 f32 C/D layout: VGPR r -> M = r (+8 for upper lane half)
#pragma unroll
  for (int mi = 0; mi < 2; ++mi) {
#pragma unroll
    for (int ni = 0; ni < 2; ++ni) {
      int col = colBase + waveN + ni * 16 + l16;
      float bv = 0.0f;
      if (ep != EP_NONE && bias) bv = bias[col];
#pragma unroll
      for (int r = 0; r < 8; ++r) {
        int row = rowBase + waveM + mi * 16 + (half ? 8 : 0) + r;
        float v = acc[mi][ni][r];
        if (ep != EP_NONE) {
          v += bv;
          if (ep == EP_BIAS_RELU)          v = fmaxf(v, 0.0f);
          else if (ep == EP_BIAS_SOFTPLUS) v = (v > 20.0f) ? v : log1pf(__expf(v));
        }
        C[(size_t)row * ldc + col] = v;
      }
    }
  }
}

// ---------------------------------------------------------------------------
// Causal depthwise conv1d (d_conv=4) + bias + SiLU.
// xz: [M, 2*D_INNER] (cols 0..1023 = pre-conv xin); out xin: [M, D_INNER].
// ---------------------------------------------------------------------------
__global__ __launch_bounds__(256) void conv_silu_kernel(
    const float* __restrict__ xz, const float* __restrict__ cw,
    const float* __restrict__ cb, float* __restrict__ xin)
{
  int idx = blockIdx.x * blockDim.x + threadIdx.x;   // over M * D_INNER
  int d = idx & (D_INNER - 1);
  int m = idx >> 10;
  int b = m / LSEQ;
  int l = m - b * LSEQ;
  float acc = cb[d];
#pragma unroll
  for (int j = 0; j < D_CONV; ++j) {
    int ll = l - (D_CONV - 1) + j;
    if (ll >= 0)
      acc = fmaf(cw[d * D_CONV + j],
                 xz[(size_t)(b * LSEQ + ll) * (2 * D_INNER) + d], acc);
  }
  xin[idx] = acc / (1.0f + __expf(-acc));            // SiLU
}

// ---------------------------------------------------------------------------
// Selective scan. One channel (b,d) per 16-lane half-wave; lane%16 = state s.
// h_s <- exp(dt*A_s)*h_s + dt*B_s*x ;  y = (sum_s h_s*C_s + x*D) * silu(z)
// Block = 256 threads = 8 waves = 16 channels. Grid = 2048/16 = 128 blocks.
// ---------------------------------------------------------------------------
__global__ __launch_bounds__(256) void scan_kernel(
    const float* __restrict__ dtf,   // [M, D_INNER] softplus(dt)
    const float* __restrict__ dbl,   // [M, 64]  (cols 32..47 = B, 48..63 = C)
    const float* __restrict__ xin,   // [M, D_INNER]
    const float* __restrict__ xz,    // [M, 2*D_INNER] (z at cols 1024..2047)
    const float* __restrict__ A_log, // [D_INNER, D_STATE]
    const float* __restrict__ Dp,    // [D_INNER]
    float* __restrict__ y)           // [M, D_INNER]
{
  const int lane = threadIdx.x & 31;
  const int half = lane >> 4;
  const int s    = lane & 15;
  const int wave = threadIdx.x >> 5;
  const int ch   = blockIdx.x * 16 + wave * 2 + half;  // 0..2047
  const int b    = ch >> 10;
  const int d    = ch & (D_INNER - 1);

  const float Av = -__expf(A_log[d * D_STATE + s]);
  const float Dv = Dp[d];
  float h = 0.0f;

  for (int l = 0; l < LSEQ; ++l) {
    const size_t m = (size_t)b * LSEQ + l;
    float dt = dtf[m * D_INNER + d];
    float xv = xin[m * D_INNER + d];
    float Bm = dbl[m * 64 + DT_RANK + s];
    float Cm = dbl[m * 64 + DT_RANK + D_STATE + s];
    float dA = __expf(dt * Av);
    h = fmaf(dA, h, dt * Bm * xv);
    float part = h * Cm;
    // reduce across the 16 states of this half-wave
    part += __shfl_xor(part, 1, 16);
    part += __shfl_xor(part, 2, 16);
    part += __shfl_xor(part, 4, 16);
    part += __shfl_xor(part, 8, 16);
    if (s == 0) {
      float yv = part + xv * Dv;
      float z  = xz[m * (2 * D_INNER) + D_INNER + d];
      yv *= z / (1.0f + __expf(-z));                 // * silu(z)
      y[m * D_INNER + d] = yv;
    }
  }
}

// ---------------------------------------------------------------------------
__global__ __launch_bounds__(256) void add_kernel(
    const float* __restrict__ a, const float* __restrict__ b,
    float* __restrict__ c, int n)
{
  int i = blockIdx.x * blockDim.x + threadIdx.x;
  if (i < n) c[i] = a[i] + b[i];
}

// ---------------------------------------------------------------------------
extern "C" void kernel_launch(void* const* d_in, const int* in_sizes, int n_in,
                              void* d_out, int out_size, void* d_ws, size_t ws_size,
                              hipStream_t stream) {
  (void)in_sizes; (void)n_in; (void)out_size; (void)ws_size;

  const float* x0   = (const float*)d_in[0];   // [B, L, D_MODEL]
  const float* ipw  = (const float*)d_in[1];   // [2, 2048, 512]
  const float* cw   = (const float*)d_in[2];   // [2, 1024, 4]
  const float* cb   = (const float*)d_in[3];   // [2, 1024]
  const float* xpw  = (const float*)d_in[4];   // [2, 64, 1024]
  const float* dpw  = (const float*)d_in[5];   // [2, 1024, 32]
  const float* dpb  = (const float*)d_in[6];   // [2, 1024]
  const float* Alog = (const float*)d_in[7];   // [2, 1024, 16]
  const float* Dp   = (const float*)d_in[8];   // [2, 1024]
  const float* opw  = (const float*)d_in[9];   // [2, 512, 1024]
  const float* w1   = (const float*)d_in[10];  // [1024, 512]
  const float* b1   = (const float*)d_in[11];  // [1024]
  const float* w2   = (const float*)d_in[12];  // [512, 1024]
  const float* b2   = (const float*)d_in[13];  // [512]
  float* out = (float*)d_out;

  // Workspace carve (floats)
  float* ws  = (float*)d_ws;
  float* xz  = ws;                          // 4096*2048
  float* xin = xz  + (size_t)MM * 2048;     // 4096*1024
  float* dbl = xin + (size_t)MM * 1024;     // 4096*64
  float* dtf = dbl + (size_t)MM * 64;       // 4096*1024
  float* yb  = dtf + (size_t)MM * 1024;     // 4096*1024
  float* xa  = yb  + (size_t)MM * 1024;     // 4096*512
  float* xb  = xa  + (size_t)MM * 512;      // 4096*512
  float* hh   = xin;                        // reuse after layers done
  float* hbuf = yb;                         // reuse after layers done

  const float* xcur = x0;
  float* louts[2] = { xa, xb };

  for (int i = 0; i < N_LAYERS; ++i) {
    const float* ipw_i  = ipw  + (size_t)i * 2 * D_INNER * D_MODEL;
    const float* cw_i   = cw   + (size_t)i * D_INNER * D_CONV;
    const float* cb_i   = cb   + (size_t)i * D_INNER;
    const float* xpw_i  = xpw  + (size_t)i * (DT_RANK + 2 * D_STATE) * D_INNER;
    const float* dpw_i  = dpw  + (size_t)i * D_INNER * DT_RANK;
    const float* dpb_i  = dpb  + (size_t)i * D_INNER;
    const float* Alog_i = Alog + (size_t)i * D_INNER * D_STATE;
    const float* Dp_i   = Dp   + (size_t)i * D_INNER;
    const float* opw_i  = opw  + (size_t)i * D_MODEL * D_INNER;

    // xz = xcur @ ipw^T                     [4096, 2048]
    gemm_wmma_f16<<<dim3(MM / 128, 2048 / 64), 256, 0, stream>>>(
        xcur, ipw_i, xz, nullptr, MM, 2048, 512, 512, 512, 2048, EP_NONE);
    // xin = silu(conv(xz[:, :1024]) + cb)
    conv_silu_kernel<<<(MM * D_INNER) / 256, 256, 0, stream>>>(xz, cw_i, cb_i, xin);
    // dbl = xin @ xpw^T                     [4096, 64]
    gemm_wmma_f16<<<dim3(MM / 128, 64 / 64), 256, 0, stream>>>(
        xin, xpw_i, dbl, nullptr, MM, 64, 1024, 1024, 1024, 64, EP_NONE);
    // dtf = softplus(dbl[:, :32] @ dpw^T + dpb)   [4096, 1024]
    gemm_wmma_f16<<<dim3(MM / 128, 1024 / 64), 256, 0, stream>>>(
        dbl, dpw_i, dtf, dpb_i, MM, 1024, 32, 64, 32, 1024, EP_BIAS_SOFTPLUS);
    // selective scan -> y (fused * silu(z))
    scan_kernel<<<128, 256, 0, stream>>>(dtf, dbl, xin, xz, Alog_i, Dp_i, yb);
    // layer out = y @ opw^T                 [4096, 512]
    gemm_wmma_f16<<<dim3(MM / 128, 512 / 64), 256, 0, stream>>>(
        yb, opw_i, louts[i], nullptr, MM, 512, 1024, 1024, 1024, 512, EP_NONE);
    xcur = louts[i];
  }

  // MLP: h = x2 + residual; out = relu(h@w1^T + b1) @ w2^T + b2
  add_kernel<<<(MM * D_MODEL + 255) / 256, 256, 0, stream>>>(xb, x0, hbuf, MM * D_MODEL);
  gemm_wmma_f16<<<dim3(MM / 128, FF_DIM / 64), 256, 0, stream>>>(
      hbuf, w1, hh, b1, MM, FF_DIM, 512, 512, 512, FF_DIM, EP_BIAS_RELU);
  gemm_wmma_f16<<<dim3(MM / 128, D_MODEL / 64), 256, 0, stream>>>(
      hh, w2, out, b2, MM, D_MODEL, FF_DIM, FF_DIM, FF_DIM, D_MODEL, EP_BIAS);
}
